// DecoderBlock_60885456388591
// MI455X (gfx1250) — matrix operand order repaired and checked by reading, not compiled, not run on hardware
//
#include <hip/hip_runtime.h>
#include <math.h>

// Problem constants (reference: B,T,S,D,H = 2,2048,2048,512,8)
#define Bc   2
#define Tc   2048
#define Sc   2048
#define Dc   512
#define Hc   8
#define HDc  64
#define MLPH 2048
#define ROWS (Bc * Tc)          // 4096
#define HEADSTRIDE ((size_t)Bc * Hc * Tc * HDc)  // 2,097,152 elems per Q/K/V matrix

typedef __attribute__((ext_vector_type(16))) __bf16 v16bf;
typedef __attribute__((ext_vector_type(8)))  float  v8f;

union Frag16 { v16bf v; unsigned int u[8]; unsigned short s[16]; };
union U4     { uint4 q; unsigned short s[8]; };

__device__ inline unsigned short f2bf(float f) {
  unsigned int x = __builtin_bit_cast(unsigned int, f);
  return (unsigned short)((x + 0x7FFFu + ((x >> 16) & 1u)) >> 16);  // RNE
}

__device__ inline v8f wmma_bf16(v16bf a, v16bf b, v8f c) {
  // D = A(16x32 bf16) * B(32x16 bf16) + C(16x16 f32)
  return __builtin_amdgcn_wmma_f32_16x16x32_bf16(false, a, false, b,
                                                 (short)0, c, false, false);
}

__device__ inline v8f vzero8() {
  v8f z;
  for (int i = 0; i < 8; ++i) z[i] = 0.0f;
  return z;
}

// Branchless GELU (tanh form): one v_exp_f32 + one v_rcp_f32.
//   tanh(y) = 1 - 2/(exp(2y)+1);  y clamped so exp never overflows.
__device__ inline float gelu_f(float v) {
  float y = 0.7978845608028654f * (v + 0.044715f * v * v * v);
  y = fminf(fmaxf(y, -15.0f), 15.0f);
  float e = __expf(2.0f * y);
  float th = 1.0f - 2.0f * __builtin_amdgcn_rcpf(e + 1.0f);
  return 0.5f * v * (1.0f + th);
}

// ---------------------------------------------------------------- converts
__global__ void f32_to_bf16_kernel(const float* __restrict__ in,
                                   unsigned short* __restrict__ out, int n) {
  int i = blockIdx.x * 256 + threadIdx.x;
  if (i < n) out[i] = f2bf(in[i]);
}

// ---------------------------------------------------------------- layernorm
// one 128-thread block per row of D=512; f32 in -> bf16 out
__global__ __launch_bounds__(128)
void ln_kernel(const float* __restrict__ x, const float* __restrict__ g,
               const float* __restrict__ bb, unsigned short* __restrict__ out) {
  __shared__ float r1[128], r2[128];
  int row = blockIdx.x, t = threadIdx.x;
  float4 xv = *(const float4*)&x[(size_t)row * Dc + t * 4];
  r1[t] = xv.x + xv.y + xv.z + xv.w;
  r2[t] = xv.x * xv.x + xv.y * xv.y + xv.z * xv.z + xv.w * xv.w;
  __syncthreads();
  for (int st = 64; st > 0; st >>= 1) {
    if (t < st) { r1[t] += r1[t + st]; r2[t] += r2[t + st]; }
    __syncthreads();
  }
  float mean = r1[0] * (1.0f / Dc);
  float var  = r2[0] * (1.0f / Dc) - mean * mean;
  float inv  = rsqrtf(var + 1e-5f);
  float xe[4] = {xv.x, xv.y, xv.z, xv.w};
#pragma unroll
  for (int i = 0; i < 4; ++i) {
    int c = t * 4 + i;
    out[(size_t)row * Dc + c] = f2bf((xe[i] - mean) * inv * g[c] + bb[c]);
  }
}

// ---------------------------------------------------------------- GEMM
// C[M,N] = A[M,K](bf16) @ Bw[K,N](bf16) + bias, fused epilogue.
// Block tile 128x128, BK=32, 256 threads = 8 waves (2x4), wave tile 64x32.
__global__ __launch_bounds__(256)
void gemm_bf16_kernel(const unsigned short* __restrict__ A,
                      const unsigned short* __restrict__ Bw,
                      const float* __restrict__ bias,
                      const float* __restrict__ resid,
                      float* __restrict__ outF,
                      unsigned short* __restrict__ outB,
                      int M, int N, int K, int act, int headScatter,
                      size_t splitStride) {
  __shared__ unsigned short ldsA[128 * 32];   // row-major [m][k]
  __shared__ unsigned short ldsB[128 * 32];   // col-major [n][k]
  const int tid = threadIdx.x;
  const int lane = tid & 31, wid = tid >> 5;
  const int half = lane >> 4, l16 = lane & 15;
  const int wm = wid >> 2, wn = wid & 3;      // 2 x 4 wave grid
  const int m0 = blockIdx.y * 128, n0 = blockIdx.x * 128;

  v8f acc[4][2];
#pragma unroll
  for (int tm = 0; tm < 4; ++tm)
#pragma unroll
    for (int tn = 0; tn < 2; ++tn) acc[tm][tn] = vzero8();

  for (int kt = 0; kt < K; kt += 32) {
    // prefetch next K tile into near caches (global_prefetch_b8)
    if (kt + 32 < K) {
      int id = tid;
      int row = id >> 2, cc = id & 3;
      __builtin_prefetch(&A[(size_t)(m0 + row) * K + kt + 32 + cc * 8], 0, 3);
      int krow = id >> 4, nc = id & 15;
      __builtin_prefetch(&Bw[(size_t)(kt + 32 + krow) * N + n0 + nc * 8], 0, 3);
    }
    __syncthreads();
#pragma unroll
    for (int j = 0; j < 2; ++j) {               // A tile: 128x32 bf16
      int id = tid + 256 * j;
      int row = id >> 2, cc = id & 3;
      *(uint4*)&ldsA[row * 32 + cc * 8] =
          *(const uint4*)&A[(size_t)(m0 + row) * K + kt + cc * 8];
    }
#pragma unroll
    for (int j = 0; j < 2; ++j) {               // B tile: 32x128 -> [n][k]
      int id = tid + 256 * j;
      int krow = id >> 4, cc = id & 15;
      U4 t;
      t.q = *(const uint4*)&Bw[(size_t)(kt + krow) * N + n0 + cc * 8];
#pragma unroll
      for (int e = 0; e < 8; ++e) ldsB[(cc * 8 + e) * 32 + krow] = t.s[e];
    }
    __syncthreads();

    Frag16 af[4];
#pragma unroll
    for (int tm = 0; tm < 4; ++tm) {
      int arow = wm * 64 + tm * 16 + l16;
#pragma unroll
      for (int i = 0; i < 4; ++i) {
        af[tm].u[i]     = *(const unsigned int*)&ldsA[arow * 32 + half * 16 + 2 * i];
        af[tm].u[4 + i] = *(const unsigned int*)&ldsA[arow * 32 + half * 16 + 8 + 2 * i];
      }
    }
#pragma unroll
    for (int tn = 0; tn < 2; ++tn) {
      Frag16 bfg;
      int bcol = wn * 32 + tn * 16 + l16;
#pragma unroll
      for (int i = 0; i < 8; ++i)
        bfg.u[i] = *(const unsigned int*)&ldsB[bcol * 32 + half * 16 + 2 * i];
#pragma unroll
      for (int tm = 0; tm < 4; ++tm)
        acc[tm][tn] = wmma_bf16(af[tm].v, bfg.v, acc[tm][tn]);
    }
  }

  // epilogue: bias (+GELU) (+residual) -> f32 and/or bf16 (optional head scatter)
#pragma unroll
  for (int tm = 0; tm < 4; ++tm)
#pragma unroll
    for (int tn = 0; tn < 2; ++tn) {
      int gcol = n0 + wn * 32 + tn * 16 + l16;
#pragma unroll
      for (int r = 0; r < 8; ++r) {
        int grow = m0 + wm * 64 + tm * 16 + r + half * 8;
        float v = acc[tm][tn][r] + bias[gcol];
        if (act == 1) v = gelu_f(v);
        if (resid) v += resid[(size_t)grow * N + gcol];
        if (outF) outF[(size_t)grow * N + gcol] = v;
        if (outB) {
          size_t idx;
          if (headScatter) {
            int split = gcol >> 9;          // which 512-col matrix (q/k/v)
            int c = gcol & 511;
            int h = c >> 6, hd = c & 63;
            int bbx = grow >> 11;           // row / T
            int tt = grow & 2047;
            idx = (size_t)split * splitStride +
                  (((size_t)bbx * Hc + h) * Tc + tt) * HDc + hd;
          } else {
            idx = (size_t)grow * N + gcol;
          }
          outB[idx] = f2bf(v);
        }
      }
    }
}

// ---------------------------------------------------------------- attention
// Q,K,V bf16 [B*H, L, 64]; per wave one 16-row Q tile, key blocks of 32.
template <bool CAUSAL>
__global__ __launch_bounds__(128)
void attn_kernel(const unsigned short* __restrict__ Q,
                 const unsigned short* __restrict__ Kd,
                 const unsigned short* __restrict__ V,
                 unsigned short* __restrict__ out, int Lq, int Lk) {
  __shared__ unsigned short Pst[4][16 * 32];
  const int tid = threadIdx.x;
  const int lane = tid & 31, wid = tid >> 5;
  const int half = lane >> 4, l16 = lane & 15;
  const int qtile = blockIdx.x * 64 + wid * 16;
  const int bh = blockIdx.y;
  const int b = bh / Hc, h = bh % Hc;
  const unsigned short* Qb = Q + (size_t)bh * Lq * HDc;
  const unsigned short* Kb = Kd + (size_t)bh * Lk * HDc;
  const unsigned short* Vb = V + (size_t)bh * Lk * HDc;

  // Q fragments (two 16x32 chunks over d)
  Frag16 qf[2];
  {
    int qrow = qtile + l16;
#pragma unroll
    for (int c = 0; c < 2; ++c) {
      const unsigned short* p = &Qb[(size_t)qrow * HDc + 32 * c];
#pragma unroll
      for (int i = 0; i < 4; ++i) {
        qf[c].u[i]     = *(const unsigned int*)&p[half * 16 + 2 * i];
        qf[c].u[4 + i] = *(const unsigned int*)&p[half * 16 + 8 + 2 * i];
      }
    }
  }

  v8f O[4];
#pragma unroll
  for (int nt = 0; nt < 4; ++nt) O[nt] = vzero8();
  float mrow[8], lrow[8];
#pragma unroll
  for (int r = 0; r < 8; ++r) { mrow[r] = -3.0e38f; lrow[r] = 0.0f; }

  const float scale = 0.125f;  // 1/sqrt(64)
  const int nkb = CAUSAL ? ((qtile + 16 + 31) >> 5) : (Lk >> 5);

  for (int kb = 0; kb < nkb; ++kb) {
    const int k0 = kb * 32;
    // prefetch next key block of K and V into near caches
    if (kb + 1 < nkb) {
      __builtin_prefetch(&Kb[(size_t)(k0 + 32 + lane) * HDc], 0, 3);
      __builtin_prefetch(&Vb[(size_t)(k0 + 32 + lane) * HDc], 0, 3);
    }
    v8f S[2];
#pragma unroll
    for (int jt = 0; jt < 2; ++jt) {
      v8f sa = vzero8();
      int key = k0 + jt * 16 + l16;
#pragma unroll
      for (int c = 0; c < 2; ++c) {            // K^T fragment: contiguous in d
        Frag16 kf;
        const unsigned short* p = &Kb[(size_t)key * HDc + 32 * c + half * 16];
#pragma unroll
        for (int i = 0; i < 8; ++i) kf.u[i] = *(const unsigned int*)&p[2 * i];
        sa = wmma_bf16(qf[c].v, kf.v, sa);
      }
      S[jt] = sa;
    }
    // scale + causal mask (only diagonal blocks need elementwise masking)
    bool diag = CAUSAL && (k0 + 31 > qtile);
#pragma unroll
    for (int jt = 0; jt < 2; ++jt) {
      int key = k0 + jt * 16 + l16;
#pragma unroll
      for (int r = 0; r < 8; ++r) {
        float s = S[jt][r] * scale;
        if (diag) {
          int qr = qtile + r + half * 8;
          if (key > qr) s = -3.0e38f;
        }
        S[jt][r] = s;
      }
    }
    // online softmax (row = one VGPR across a 16-lane half-group)
    float alpha[8];
#pragma unroll
    for (int r = 0; r < 8; ++r) {
      float rm = fmaxf(S[0][r], S[1][r]);
#pragma unroll
      for (int d = 1; d < 16; d <<= 1) rm = fmaxf(rm, __shfl_xor(rm, d));
      float mn = fmaxf(mrow[r], rm);
      alpha[r] = __expf(mrow[r] - mn);
      mrow[r] = mn;
      float p0 = __expf(S[0][r] - mn);
      float p1 = __expf(S[1][r] - mn);
      S[0][r] = p0; S[1][r] = p1;
      float rs = p0 + p1;
#pragma unroll
      for (int d = 1; d < 16; d <<= 1) rs += __shfl_xor(rs, d);
      lrow[r] = lrow[r] * alpha[r] + rs;
    }
#pragma unroll
    for (int nt = 0; nt < 4; ++nt)
#pragma unroll
      for (int r = 0; r < 8; ++r) O[nt][r] *= alpha[r];

    // restage P (C-layout f32) -> LDS -> (A-layout bf16); wave-private tile
    unsigned short* pw = Pst[wid];
#pragma unroll
    for (int jt = 0; jt < 2; ++jt)
#pragma unroll
      for (int r = 0; r < 8; ++r)
        pw[(r + half * 8) * 32 + jt * 16 + l16] = f2bf(S[jt][r]);
    asm volatile("s_wait_dscnt 0x0" ::: "memory");
    Frag16 pf;
#pragma unroll
    for (int i = 0; i < 4; ++i) {
      pf.u[i]     = *(const unsigned int*)&pw[l16 * 32 + half * 16 + 2 * i];
      pf.u[4 + i] = *(const unsigned int*)&pw[l16 * 32 + half * 16 + 8 + 2 * i];
    }
    // O += P @ V  (V fragment gathered along keys)
#pragma unroll
    for (int nt = 0; nt < 4; ++nt) {
      Frag16 vf;
      int d = nt * 16 + l16;
#pragma unroll
      for (int i = 0; i < 8; ++i) {
        vf.s[2 * i]     = Vb[(size_t)(k0 + half * 16 + 2 * i) * HDc + d];
        vf.s[2 * i + 1] = Vb[(size_t)(k0 + half * 16 + 2 * i + 1) * HDc + d];
      }
      O[nt] = wmma_bf16(pf.v, vf.v, O[nt]);
    }
  }

  // write merged heads: [B*Lq, D] bf16 (one v_rcp per row instead of 4 divides)
  float inv_l[8];
#pragma unroll
  for (int r = 0; r < 8; ++r) inv_l[r] = __builtin_amdgcn_rcpf(lrow[r]);
#pragma unroll
  for (int nt = 0; nt < 4; ++nt)
#pragma unroll
    for (int r = 0; r < 8; ++r) {
      int row = qtile + r + half * 8;
      float v = O[nt][r] * inv_l[r];
      out[((size_t)(b * Lq + row)) * Dc + h * HDc + nt * 16 + l16] = f2bf(v);
    }
}

// ---------------------------------------------------------------- launch
extern "C" void kernel_launch(void* const* d_in, const int* in_sizes, int n_in,
                              void* d_out, int out_size, void* d_ws, size_t ws_size,
                              hipStream_t stream) {
  (void)in_sizes; (void)n_in; (void)out_size; (void)ws_size;
  const float* x    = (const float*)d_in[0];
  const float* enc  = (const float*)d_in[1];
  const float* qkvw = (const float*)d_in[4];
  const float* qkvb = (const float*)d_in[5];
  const float* saw  = (const float*)d_in[6];
  const float* sab  = (const float*)d_in[7];
  const float* ln1g = (const float*)d_in[8];
  const float* ln1b = (const float*)d_in[9];
  const float* qw   = (const float*)d_in[10];
  const float* qb   = (const float*)d_in[11];
  const float* kvw  = (const float*)d_in[12];
  const float* kvb  = (const float*)d_in[13];
  const float* caw  = (const float*)d_in[14];
  const float* cab  = (const float*)d_in[15];
  const float* ln2g = (const float*)d_in[16];
  const float* ln2b = (const float*)d_in[17];
  const float* m1w  = (const float*)d_in[18];
  const float* m1b  = (const float*)d_in[19];
  const float* m2w  = (const float*)d_in[20];
  const float* m2b  = (const float*)d_in[21];
  const float* ln3g = (const float*)d_in[22];
  const float* ln3b = (const float*)d_in[23];
  float* outp = (float*)d_out;

  char* ws = (char*)d_ws;
  size_t off = 0;
  auto alloc = [&](size_t bytes) -> char* {
    char* p = ws + off;
    off += (bytes + 255) & ~(size_t)255;
    return p;
  };
  unsigned short* Wqkv = (unsigned short*)alloc((size_t)Dc * 3 * Dc * 2);
  unsigned short* Wsa  = (unsigned short*)alloc((size_t)Dc * Dc * 2);
  unsigned short* Wq   = (unsigned short*)alloc((size_t)Dc * Dc * 2);
  unsigned short* Wkv  = (unsigned short*)alloc((size_t)Dc * 2 * Dc * 2);
  unsigned short* Wca  = (unsigned short*)alloc((size_t)Dc * Dc * 2);
  unsigned short* Wm1  = (unsigned short*)alloc((size_t)Dc * MLPH * 2);
  unsigned short* Wm2  = (unsigned short*)alloc((size_t)MLPH * Dc * 2);
  float*          xcur = (float*)alloc((size_t)ROWS * Dc * 4);
  unsigned short* lnout  = (unsigned short*)alloc((size_t)ROWS * Dc * 2);
  unsigned short* qkvh   = (unsigned short*)alloc(3 * HEADSTRIDE * 2); // q,k,v
  unsigned short* qch    = (unsigned short*)alloc(HEADSTRIDE * 2);
  unsigned short* kvch   = (unsigned short*)alloc(2 * HEADSTRIDE * 2); // kc,vc
  unsigned short* encbf  = (unsigned short*)alloc((size_t)ROWS * Dc * 2);
  unsigned short* attout = (unsigned short*)alloc((size_t)ROWS * Dc * 2);
  unsigned short* hidden = (unsigned short*)alloc((size_t)ROWS * MLPH * 2);

  auto cvt = [&](const float* src, unsigned short* dst, int n) {
    f32_to_bf16_kernel<<<(n + 255) / 256, 256, 0, stream>>>(src, dst, n);
  };
  cvt(qkvw, Wqkv, Dc * 3 * Dc);
  cvt(saw,  Wsa,  Dc * Dc);
  cvt(qw,   Wq,   Dc * Dc);
  cvt(kvw,  Wkv,  Dc * 2 * Dc);
  cvt(caw,  Wca,  Dc * Dc);
  cvt(m1w,  Wm1,  Dc * MLPH);
  cvt(m2w,  Wm2,  MLPH * Dc);
  cvt(enc,  encbf, ROWS * Dc);

  hipMemcpyAsync(xcur, x, (size_t)ROWS * Dc * 4, hipMemcpyDeviceToDevice, stream);

  // ---- self-attention
  ln_kernel<<<ROWS, 128, 0, stream>>>(xcur, ln1g, ln1b, lnout);
  gemm_bf16_kernel<<<dim3(3 * Dc / 128, ROWS / 128), 256, 0, stream>>>(
      lnout, Wqkv, qkvb, nullptr, nullptr, qkvh, ROWS, 3 * Dc, Dc, 0, 1, HEADSTRIDE);
  attn_kernel<true><<<dim3(Tc / 64, Bc * Hc), 128, 0, stream>>>(
      qkvh, qkvh + HEADSTRIDE, qkvh + 2 * HEADSTRIDE, attout, Tc, Tc);
  gemm_bf16_kernel<<<dim3(Dc / 128, ROWS / 128), 256, 0, stream>>>(
      attout, Wsa, sab, xcur, xcur, nullptr, ROWS, Dc, Dc, 0, 0, 0);

  // ---- cross-attention
  ln_kernel<<<ROWS, 128, 0, stream>>>(xcur, ln2g, ln2b, lnout);
  gemm_bf16_kernel<<<dim3(Dc / 128, ROWS / 128), 256, 0, stream>>>(
      lnout, Wq, qb, nullptr, nullptr, qch, ROWS, Dc, Dc, 0, 1, HEADSTRIDE);
  gemm_bf16_kernel<<<dim3(2 * Dc / 128, ROWS / 128), 256, 0, stream>>>(
      encbf, Wkv, kvb, nullptr, nullptr, kvch, ROWS, 2 * Dc, Dc, 0, 1, HEADSTRIDE);
  attn_kernel<false><<<dim3(Tc / 64, Bc * Hc), 128, 0, stream>>>(
      qch, kvch, kvch + HEADSTRIDE, attout, Tc, Sc);
  gemm_bf16_kernel<<<dim3(Dc / 128, ROWS / 128), 256, 0, stream>>>(
      attout, Wca, cab, xcur, xcur, nullptr, ROWS, Dc, Dc, 0, 0, 0);

  // ---- MLP
  ln_kernel<<<ROWS, 128, 0, stream>>>(xcur, ln3g, ln3b, lnout);
  gemm_bf16_kernel<<<dim3(MLPH / 128, ROWS / 128), 256, 0, stream>>>(
      lnout, Wm1, m1b, nullptr, nullptr, hidden, ROWS, MLPH, Dc, 1, 0, 0);
  gemm_bf16_kernel<<<dim3(Dc / 128, ROWS / 128), 256, 0, stream>>>(
      hidden, Wm2, m2b, xcur, outp, nullptr, ROWS, Dc, MLPH, 0, 0, 0);
}
